// InGramRelationLayer_72533407695107
// MI455X (gfx1250) — matrix operand order repaired and checked by reading, not compiled, not run on hardware
//
#include <hip/hip_runtime.h>
#include <hip/hip_bf16.h>

#define NREL     20000
#define NEDGE    400000
#define DIMIN    512
#define DIMOUT   512
#define NHEAD    8
#define DHID     64
#define NBIN     10
#define NFRAG    (3 * 32 * 16)   // mats * n-tiles * k-steps

typedef __attribute__((ext_vector_type(16))) _Float16 v16h;
typedef __attribute__((ext_vector_type(8)))  float    v8f;

__device__ __forceinline__ float lrelu(float x) { return x > 0.f ? x : 0.2f * x; }

// ---------------------------------------------------------------------------
// init helpers
// ---------------------------------------------------------------------------
__global__ void zero_i_kernel(int* __restrict__ p, int n) {
    int i = blockIdx.x * blockDim.x + threadIdx.x;
    if (i < n) p[i] = 0;
}

// ---------------------------------------------------------------------------
// CSR build: counts -> scan -> fill
// ---------------------------------------------------------------------------
__global__ __launch_bounds__(256) void ingram_count(const int* __restrict__ trip,
                                                    int* __restrict__ counts) {
    int e = blockIdx.x * 256 + threadIdx.x;
    if (e < NEDGE) atomicAdd(&counts[trip[e * 3 + 0]], 1);
}

// single-block exclusive scan of counts[NREL] -> offsets[NREL+1]
__global__ __launch_bounds__(1024) void ingram_scan(const int* __restrict__ counts,
                                                    int* __restrict__ offsets) {
    __shared__ int warpsums[32];
    __shared__ int carry_s;
    const int tid = threadIdx.x;
    const int lane = tid & 31;
    const int wid = tid >> 5;
    if (tid == 0) { carry_s = 0; offsets[0] = 0; }
    __syncthreads();
    for (int base = 0; base < NREL; base += 1024) {
        const int i = base + tid;
        int v = (i < NREL) ? counts[i] : 0;
#pragma unroll
        for (int d = 1; d < 32; d <<= 1) {
            int t = __shfl_up(v, d, 32);
            if (lane >= d) v += t;
        }
        if (lane == 31) warpsums[wid] = v;
        __syncthreads();
        if (wid == 0) {
            int w = warpsums[lane];
#pragma unroll
            for (int d = 1; d < 32; d <<= 1) {
                int t = __shfl_up(w, d, 32);
                if (lane >= d) w += t;
            }
            warpsums[lane] = w;
        }
        __syncthreads();
        const int add = (wid > 0) ? warpsums[wid - 1] : 0;
        const int carry = carry_s;
        if (i < NREL) offsets[i + 1] = carry + v + add;
        __syncthreads();
        if (tid == 0) carry_s = carry + warpsums[31];
        __syncthreads();
    }
}

__global__ __launch_bounds__(256) void ingram_copy_cursor(const int* __restrict__ offsets,
                                                          int* __restrict__ cursor) {
    int i = blockIdx.x * 256 + threadIdx.x;
    if (i < NREL) cursor[i] = offsets[i];
}

__global__ __launch_bounds__(256) void ingram_fill(const int* __restrict__ trip,
                                                   int* __restrict__ cursor,
                                                   int* __restrict__ eid) {
    int e = blockIdx.x * 256 + threadIdx.x;
    if (e >= NEDGE) return;
    int pos = atomicAdd(&cursor[trip[e * 3 + 0]], 1);
    eid[pos] = e;
}

// ---------------------------------------------------------------------------
// Pack the three weight matrices into WMMA B-fragment layout (f16).
// fragment f = (mat*32 + ntile)*16 + kstep ; per lane 16 contiguous halves.
//   mat 0: attn_w rows [0,512)    (W_top)
//   mat 1: attn_w rows [512,1024) (W_bot)
//   mat 2: aggr_w
// ---------------------------------------------------------------------------
__global__ __launch_bounds__(256) void ingram_pack_b(
    const float* __restrict__ attn_w, const float* __restrict__ aggr_w,
    _Float16* __restrict__ Bpk)
{
    const int f = blockIdx.x * 8 + (threadIdx.x >> 5);
    if (f >= NFRAG) return;
    const int lane  = threadIdx.x & 31;
    const int mat   = f >> 9;          // /512
    const int ntile = (f >> 4) & 31;
    const int kstep = f & 15;
    const float* Bp = (mat == 2) ? aggr_w : attn_w;
    const int koff  = (mat == 1) ? 512 : 0;
    const int n  = ntile * 16 + (lane & 15);
    const int hi = lane >> 4;
    const int k0 = kstep * 32;
    v16h val;
#pragma unroll
    for (int v = 0; v < 8; ++v) {
        int k = k0 + hi * 16 + v * 2 + koff;
        val[v * 2 + 0] = (_Float16)Bp[(size_t)(k + 0) * DIMOUT + n];
        val[v * 2 + 1] = (_Float16)Bp[(size_t)(k + 1) * DIMOUT + n];
    }
    *(v16h*)(Bpk + ((size_t)f * 32 + lane) * 16) = val;
}

// ---------------------------------------------------------------------------
// Relation-level GEMMs on v_wmma_f32_16x16x32_f16.
//   blockIdx.y = mat (0:P_h, 1:P_t, 2:M), blockIdx.x = 16-row tile.
//   8 waves/block, each wave owns 4 N-tiles; A fragment reused 4x,
//   B fragments are single contiguous v16h loads from the packed buffer.
// ---------------------------------------------------------------------------
__global__ __launch_bounds__(256) void ingram_gemm_proj(
    const float* __restrict__ emb, const _Float16* __restrict__ Bpk,
    const float* __restrict__ aggr_b,
    float* __restrict__ Ph, float* __restrict__ Pt, float* __restrict__ Mb)
{
    const int lane = threadIdx.x & 31;
    const int wave = threadIdx.x >> 5;
    const int r    = lane & 15;
    const int hi   = lane >> 4;
    const int mbase = blockIdx.x * 16;
    const int mat   = blockIdx.y;

    v8f acc[4];
#pragma unroll
    for (int j = 0; j < 4; ++j)
#pragma unroll
        for (int q = 0; q < 8; ++q) acc[j][q] = 0.f;

    const float* arow = emb + (size_t)(mbase + r) * DIMIN;

    for (int k0 = 0; k0 < DIMIN; k0 += 32) {
        const int kstep = k0 >> 5;
        v16h a;
#pragma unroll
        for (int v = 0; v < 8; ++v) {
            int kb = k0 + hi * 16 + v * 2;
            a[v * 2 + 0] = (_Float16)arow[kb + 0];
            a[v * 2 + 1] = (_Float16)arow[kb + 1];
        }
#pragma unroll
        for (int j = 0; j < 4; ++j) {
            const int f = ((mat * 32 + (wave * 4 + j)) * 16 + kstep);
            v16h b = *(const v16h*)(Bpk + ((size_t)f * 32 + lane) * 16);
            acc[j] = __builtin_amdgcn_wmma_f32_16x16x32_f16(
                false, a, false, b, (short)0, acc[j], false, false);
        }
    }

    float* Out = (mat == 0) ? Ph : ((mat == 1) ? Pt : Mb);
#pragma unroll
    for (int j = 0; j < 4; ++j) {
        const int n = (wave * 4 + j) * 16 + r;
        const float bias = (mat == 2) ? aggr_b[n] : 0.f;
#pragma unroll
        for (int v = 0; v < 8; ++v) {
            const int row = mbase + v + hi * 8;
            Out[(size_t)row * DIMOUT + n] = acc[j][v] + bias;
        }
    }
}

// ---------------------------------------------------------------------------
// One wave per head-relation: walk its edge segment in ascending edge-id
// order (deterministic), online-softmax over per-head logits, accumulate
// w * M[tail] in registers, write the output row exactly once. No atomics.
// lane owns 16 contiguous channels; head = lane>>2.
// ---------------------------------------------------------------------------
__global__ __launch_bounds__(256) void ingram_aggregate(
    const float* __restrict__ Ph, const float* __restrict__ Pt,
    const float* __restrict__ Mb, const float* __restrict__ pb,
    const float* __restrict__ av, const float* __restrict__ abin,
    const int* __restrict__ trip, const int* __restrict__ offsets,
    const int* __restrict__ eid, float* __restrict__ out)
{
    const int r = blockIdx.x * 8 + (threadIdx.x >> 5);
    if (r >= NREL) return;
    const int lane = threadIdx.x & 31;
    const int h = lane >> 2;
    const int p0 = offsets[r], p1 = offsets[r + 1];

    const float4* ph4 = (const float4*)(Ph + (size_t)r * DIMOUT) + lane * 4;
    const float4* pb4 = (const float4*)pb + lane * 4;
    const float4* av4 = (const float4*)av + lane * 4;
    float4 pre[4], vv[4];
#pragma unroll
    for (int q = 0; q < 4; ++q) {
        float4 a = ph4[q], b = pb4[q];
        pre[q].x = a.x + b.x; pre[q].y = a.y + b.y;
        pre[q].z = a.z + b.z; pre[q].w = a.w + b.w;
        vv[q] = av4[q];
    }

    float m = -INFINITY, s = 0.f;
    float4 acc[4];
#pragma unroll
    for (int q = 0; q < 4; ++q) { acc[q].x = acc[q].y = acc[q].z = acc[q].w = 0.f; }

    int last = -1;
    for (int step = p0; step < p1; ++step) {
        // deterministic traversal: smallest edge id greater than `last`
        int cur = 0x7fffffff;
        for (int j = p0; j < p1; ++j) {
            int id = eid[j];
            if (id > last && id < cur) cur = id;
        }
        last = cur;
        const int tail = trip[cur * 3 + 1];
        const int bin  = trip[cur * 3 + 2];

        const float4* t4 = (const float4*)(Pt + (size_t)tail * DIMOUT) + lane * 4;
        const float4* m4 = (const float4*)(Mb + (size_t)tail * DIMOUT) + lane * 4;
        float4 mv[4];
        float part = 0.f;
#pragma unroll
        for (int q = 0; q < 4; ++q) {
            float4 t = t4[q];
            mv[q] = m4[q];
            part += lrelu(pre[q].x + t.x) * vv[q].x;
            part += lrelu(pre[q].y + t.y) * vv[q].y;
            part += lrelu(pre[q].z + t.z) * vv[q].z;
            part += lrelu(pre[q].w + t.w) * vv[q].w;
        }
        part += __shfl_xor(part, 1);
        part += __shfl_xor(part, 2);

        const float logit = part + abin[bin * NHEAD + h];
        const float mn = fmaxf(m, logit);
        const float sc = __expf(m - mn);      // 0 on first edge (m = -inf)
        const float w  = __expf(logit - mn);
        s = s * sc + w;
#pragma unroll
        for (int q = 0; q < 4; ++q) {
            acc[q].x = fmaf(acc[q].x, sc, w * mv[q].x);
            acc[q].y = fmaf(acc[q].y, sc, w * mv[q].y);
            acc[q].z = fmaf(acc[q].z, sc, w * mv[q].z);
            acc[q].w = fmaf(acc[q].w, sc, w * mv[q].w);
        }
        m = mn;
    }

    const float inv = 1.f / (s + 1e-16f);     // empty segment -> acc = 0 -> out = 0
    float4* orow = (float4*)(out + (size_t)r * DIMOUT) + lane * 4;
#pragma unroll
    for (int q = 0; q < 4; ++q) {
        float4 o;
        o.x = acc[q].x * inv; o.y = acc[q].y * inv;
        o.z = acc[q].z * inv; o.w = acc[q].w * inv;
        orow[q] = o;
    }
}

// ---------------------------------------------------------------------------
extern "C" void kernel_launch(void* const* d_in, const int* in_sizes, int n_in,
                              void* d_out, int out_size, void* d_ws, size_t ws_size,
                              hipStream_t stream) {
    const float* emb    = (const float*)d_in[0];  // [20000,512]
    const float* attn_w = (const float*)d_in[1];  // [1024,512]
    const float* attn_b = (const float*)d_in[2];  // [512]
    const float* abin   = (const float*)d_in[3];  // [10,8,1]
    const float* avec   = (const float*)d_in[4];  // [1,8,64]
    const float* aggr_w = (const float*)d_in[5];  // [512,512]
    const float* aggr_b = (const float*)d_in[6];  // [512]
    const int*   trip   = (const int*)d_in[7];    // [E,3]
    float* out = (float*)d_out;

    // workspace layout (4-byte units)
    float* ws = (float*)d_ws;
    const size_t PROJ = (size_t)NREL * DIMOUT;           // 10,240,000
    float*     Ph  = ws;
    float*     Pt  = Ph + PROJ;
    float*     Mb  = Pt + PROJ;
    _Float16*  Bpk = (_Float16*)(Mb + PROJ);             // NFRAG*32*16 halves
    int*       eid     = (int*)(Mb + PROJ + (size_t)NFRAG * 32 * 16 / 2);
    int*       counts  = eid + NEDGE;
    int*       offsets = counts + NREL;                  // NREL+1
    int*       cursor  = offsets + NREL + 1;

    // CSR build
    zero_i_kernel<<<(NREL + 255) / 256, 256, 0, stream>>>(counts, NREL);
    ingram_count<<<(NEDGE + 255) / 256, 256, 0, stream>>>(trip, counts);
    ingram_scan<<<1, 1024, 0, stream>>>(counts, offsets);
    ingram_copy_cursor<<<(NREL + 255) / 256, 256, 0, stream>>>(offsets, cursor);
    ingram_fill<<<(NEDGE + 255) / 256, 256, 0, stream>>>(trip, cursor, eid);

    // weights -> WMMA fragment layout (f16)
    ingram_pack_b<<<NFRAG / 8, 256, 0, stream>>>(attn_w, aggr_w, Bpk);

    // three relation-level GEMMs on the WMMA path
    ingram_gemm_proj<<<dim3(NREL / 16, 3), 256, 0, stream>>>(
        emb, Bpk, aggr_b, Ph, Pt, Mb);

    // per-relation online-softmax aggregation (atomic-free, writes out once)
    ingram_aggregate<<<NREL / 8, 256, 0, stream>>>(
        Ph, Pt, Mb, attn_b, avec, abin, trip, offsets, eid, out);
}
